// SwinTransformerBlock_67551245631940
// MI455X (gfx1250) — matrix operand order repaired and checked by reading, not compiled, not run on hardware
//
#include <hip/hip_runtime.h>
#include <hip/hip_bf16.h>
#include <stdint.h>

typedef __attribute__((ext_vector_type(16))) _Float16 v16h;
typedef __attribute__((ext_vector_type(8)))  _Float16 v8h;
typedef __attribute__((ext_vector_type(8)))  float    v8f;

#define WMMA_F16(a, b, c) \
  __builtin_amdgcn_wmma_f32_16x16x32_f16(false, (a), false, (b), (short)0, (c), false, false)

// ---------------- problem constants ----------------
static constexpr int Bb  = 8;
static constexpr int Hh  = 256;
static constexpr int Ww  = 256;
static constexpr int Cc  = 48;
static constexpr int NH  = 3;
static constexpr int WS  = 8;
static constexpr int SS  = 4;
static constexpr int Nn  = 64;    // tokens / window
static constexpr int HID = 192;
static constexpr int NWIN = (Hh / WS) * (Ww / WS);       // 1024 windows / batch
static constexpr int NTOK = Bb * Hh * Ww;                // 524288 tokens

// ---------------- WMMA fragment helpers (wave32 layouts, CDNA5 ISA) ----
// A fragment 16x32 f16 from LDS laid out [m][ldk] (ldk multiple of 8 halfs)
__device__ __forceinline__ v16h ldA(const _Float16* p, int ldk) {
  const int lane = threadIdx.x & 31;
  const _Float16* row = p + (lane & 15) * ldk + ((lane >> 4) << 3);
  const v8h lo = *(const v8h*)(row);
  const v8h hi = *(const v8h*)(row + 16);
  v16h a;
#pragma unroll
  for (int i = 0; i < 8; ++i) { a[i] = lo[i]; a[i + 8] = hi[i]; }
  return a;
}

// B fragment 32x16 f16 from LDS laid out [n][ldk] (k contiguous per output column)
__device__ __forceinline__ v16h ldB(const _Float16* p, int ldk) {
  const int lane = threadIdx.x & 31;
  const _Float16* row = p + (lane & 15) * ldk + ((lane >> 4) << 3);
  const v8h lo = *(const v8h*)(row);
  const v8h hi = *(const v8h*)(row + 16);
  v16h b;
#pragma unroll
  for (int i = 0; i < 8; ++i) { b[i] = lo[i]; b[i + 8] = hi[i]; }
  return b;
}

__device__ __forceinline__ int region9(int pr, int pc) {
  const int rh = pr < (Hh - WS) ? 0 : (pr < (Hh - SS) ? 1 : 2);
  const int rc = pc < (Ww - WS) ? 0 : (pc < (Ww - SS) ? 1 : 2);
  return rh * 3 + rc;
}

// ---------------- dynamic LDS layouts ----------------
// attention kernel
static constexpr int OFF_XW  = 0;                       // 64x64 f16          8192
static constexpr int OFF_Q   = 8192;                    // 3x64x32 f16       12288
static constexpr int OFF_K   = 20480;                   // 3x64x32 f16       12288
static constexpr int OFF_VT  = 32768;                   // 3x16x64 f16        6144
static constexpr int OFF_W   = 38912;                   // 144x64 f16        18432
static constexpr int OFF_P   = 57344;                   // 3x64x64 f16       24576
static constexpr int OFF_AO  = 81920;                   // 64x48 f32         12288
static constexpr int OFF_A2  = 94208;                   // 64x64 f16          8192
static constexpr int OFF_RPB = 102400;                  // 225x3 f32          2700
static constexpr int OFF_SEA = 105104;                  // 48 f32
static constexpr int OFF_SEB = 105296;                  // 48 f32
static constexpr int OFF_SE2 = 105488;                  // 3 f32
static constexpr int SMEM_ATTN = 105600;

// mlp kernel
static constexpr int OFF_MA  = 0;                       // 64x64 f16          8192
static constexpr int OFF_W1  = 8192;                    // 192x64 f16        24576
static constexpr int OFF_HB  = 32768;                   // 64x192 f16        24576
static constexpr int OFF_W2  = 57344;                   // 48x192 f16        18432
static constexpr int SMEM_MLP = 75776;

// =====================================================================
// Kernel 1: LN1 + cyclic shift + window partition + QKV + windowed
// attention (rpb + shift mask) + softmax + SE branch + proj + residual.
// One block (4 wave32) per window.  xmid = x + attn_out  (written to d_out)
// =====================================================================
__global__ void __launch_bounds__(128)
swin_attn_kernel(const float* __restrict__ x,
                 const float* __restrict__ g1, const float* __restrict__ b1,
                 const float* __restrict__ qkv_w, const float* __restrict__ qkv_b,
                 const float* __restrict__ rpb_table, const float* __restrict__ wf,
                 const float* __restrict__ se_w1, const float* __restrict__ se_b1,
                 const float* __restrict__ se_w2, const float* __restrict__ se_b2,
                 const float* __restrict__ proj_w, const float* __restrict__ proj_b,
                 float* __restrict__ xmid) {
  extern __shared__ char smem[];
  _Float16* XW = (_Float16*)(smem + OFF_XW);
  _Float16* Qs = (_Float16*)(smem + OFF_Q);
  _Float16* Ks = (_Float16*)(smem + OFF_K);
  _Float16* VT = (_Float16*)(smem + OFF_VT);
  _Float16* WB = (_Float16*)(smem + OFF_W);
  _Float16* Ps = (_Float16*)(smem + OFF_P);
  float*    AO = (float*)(smem + OFF_AO);
  _Float16* A2 = (_Float16*)(smem + OFF_A2);
  float*    RPB = (float*)(smem + OFF_RPB);
  float*    SEA = (float*)(smem + OFF_SEA);
  float*    SEB = (float*)(smem + OFF_SEB);
  float*    SE2 = (float*)(smem + OFF_SE2);

  const int tid  = threadIdx.x;
  const int wave = tid >> 5;
  const int lane = tid & 31;
  const int blk  = blockIdx.x;
  const int bat  = blk >> 10;            // / 1024 windows per batch
  const int wi   = blk & 1023;
  const int wh   = wi >> 5;
  const int ww   = wi & 31;

  // hint the data mover at weights (global_prefetch_b8)
  if (tid == 0) { __builtin_prefetch(qkv_w, 0, 1); __builtin_prefetch(proj_w, 0, 1); }

  // ---- stage: zero Q/K pad, stage qkv weights (f16, K padded 48->64), rpb ----
  for (int i = tid; i < 3 * 64 * 32; i += 128) { Qs[i] = (_Float16)0.f; Ks[i] = (_Float16)0.f; }
  for (int i = tid; i < 144 * 64; i += 128) {
    const int o = i >> 6, k = i & 63;
    WB[i] = (_Float16)(k < Cc ? qkv_w[o * Cc + k] : 0.f);
  }
  for (int i = tid; i < 225 * 3; i += 128) RPB[i] = rpb_table[i];

  // ---- LN1 + shift + gather window into XW (f16, [64][64], K padded) ----
  if (tid < 64) {
    const int t = tid;
    const int rs = ((wh << 3) + (t >> 3) + SS) & (Hh - 1);
    const int cs = ((ww << 3) + (t & 7) + SS) & (Ww - 1);
    const float* row = x + ((size_t)bat * (Hh * Ww) + (size_t)rs * Ww + cs) * Cc;
    float v[Cc];
    float mu = 0.f;
#pragma unroll
    for (int c = 0; c < Cc; ++c) { v[c] = row[c]; mu += v[c]; }
    mu *= (1.f / Cc);
    float s2 = 0.f;
#pragma unroll
    for (int c = 0; c < Cc; ++c) { const float d = v[c] - mu; s2 += d * d; }
    const float inv = rsqrtf(s2 * (1.f / Cc) + 1e-5f);
#pragma unroll
    for (int c = 0; c < Cc; ++c) XW[t * 64 + c] = (_Float16)((v[c] - mu) * inv * g1[c] + b1[c]);
#pragma unroll
    for (int c = Cc; c < 64; ++c) XW[t * 64 + c] = (_Float16)0.f;
  }
  __syncthreads();

  // ---- QKV GEMM: [64x64] x [64x144] -> route to Q(scaled)/K/V^T in LDS ----
  {
    const int mt = wave;                 // one M-tile row per wave
    const float scale = 0.25f;           // hd^-0.5, hd = 16
    const int n = lane & 15, mb = (lane >> 4) * 8;
    for (int nt = 0; nt < 9; ++nt) {
      v8f acc = {};
#pragma unroll
      for (int ks = 0; ks < 2; ++ks) {
        const v16h a = ldA(XW + (mt * 16) * 64 + ks * 32, 64);
        const v16h b = ldB(WB + (nt * 16) * 64 + ks * 32, 64);
        acc = WMMA_F16(a, b, acc);
      }
      const int ng = nt * 16 + n;
      const float bias = qkv_b[ng];
#pragma unroll
      for (int r = 0; r < 8; ++r) {
        const int m = mt * 16 + mb + r;
        const float vv = acc[r] + bias;
        if (ng < 48) {
          const int h = ng >> 4, d = ng & 15;
          Qs[(h * 64 + m) * 32 + d] = (_Float16)(vv * scale);
        } else if (ng < 96) {
          const int h = (ng - 48) >> 4, d = ng & 15;
          Ks[(h * 64 + m) * 32 + d] = (_Float16)vv;
        } else {
          const int h = (ng - 96) >> 4, d = ng & 15;
          VT[(h * 16 + d) * 64 + m] = (_Float16)vv;     // V transposed: [dim][token]
        }
      }
    }
  }
  __syncthreads();

  // ---- attention scores + rpb + mask + register softmax -> P (f16) ----
  for (int rb = wave; rb < 12; rb += 4) {              // rb = head*4 + mtile
    const int h = rb >> 2, mt = rb & 3;
    const int n = lane & 15, mb = (lane >> 4) * 8;
    float vals[4][8];
    const v16h a = ldA(Qs + (h * 64 + mt * 16) * 32, 32);
#pragma unroll
    for (int nt = 0; nt < 4; ++nt) {
      const v16h b = ldB(Ks + (h * 64 + nt * 16) * 32, 32);
      v8f acc = {};
      acc = WMMA_F16(a, b, acc);
      const int j = nt * 16 + n;
      const int rj = region9((wh << 3) + (j >> 3), (ww << 3) + (j & 7));
#pragma unroll
      for (int r = 0; r < 8; ++r) {
        const int i = mt * 16 + mb + r;
        const int dy = (i >> 3) - (j >> 3), dx = (i & 7) - (j & 7);
        const float rp = RPB[((dy + 7) * 15 + (dx + 7)) * 3 + h];
        const int ri = region9((wh << 3) + (i >> 3), (ww << 3) + (i & 7));
        vals[nt][r] = acc[r] + rp + ((ri != rj) ? -100.f : 0.f);
      }
    }
    // row softmax: rows live per 16-lane half, columns across 4 tiles
#pragma unroll
    for (int r = 0; r < 8; ++r) {
      float mx = fmaxf(fmaxf(vals[0][r], vals[1][r]), fmaxf(vals[2][r], vals[3][r]));
#pragma unroll
      for (int off = 8; off >= 1; off >>= 1) mx = fmaxf(mx, __shfl_xor(mx, off, 32));
      float s = 0.f;
#pragma unroll
      for (int nt = 0; nt < 4; ++nt) { vals[nt][r] = __expf(vals[nt][r] - mx); s += vals[nt][r]; }
#pragma unroll
      for (int off = 8; off >= 1; off >>= 1) s += __shfl_xor(s, off, 32);
      const float rs = 1.f / s;
#pragma unroll
      for (int nt = 0; nt < 4; ++nt) {
        const int i = mt * 16 + mb + r;
        const int j = nt * 16 + n;
        Ps[(h * 64 + i) * 64 + j] = (_Float16)(vals[nt][r] * rs);
      }
    }
  }
  __syncthreads();

  // ---- P @ V : [64x64] x [64x16] per head -> AO f32 [64][48] ----
  for (int t = wave; t < 12; t += 4) {
    const int h = t >> 2, mt = t & 3;
    v8f acc = {};
#pragma unroll
    for (int ks = 0; ks < 2; ++ks) {
      const v16h a = ldA(Ps + (h * 64 + mt * 16) * 64 + ks * 32, 64);
      const v16h b = ldB(VT + (h * 16) * 64 + ks * 32, 64);
      acc = WMMA_F16(a, b, acc);
    }
    const int n = lane & 15, mb = (lane >> 4) * 8;
#pragma unroll
    for (int r = 0; r < 8; ++r) AO[(mt * 16 + mb + r) * Cc + h * 16 + n] = acc[r];
  }
  __syncthreads();

  // ---- SE branch: squeeze over tokens, 48->3 relu, 3->48 sigmoid ----
  if (tid < Cc) {
    float s = 0.f;
#pragma unroll 8
    for (int m = 0; m < 64; ++m) s += (float)XW[m * 64 + tid];
    SEA[tid] = s * (1.f / 64.f);
  }
  __syncthreads();
  if (tid < 3) {
    float s = se_b1[tid];
#pragma unroll
    for (int c = 0; c < Cc; ++c) s += se_w1[tid * Cc + c] * SEA[c];
    SE2[tid] = fmaxf(s, 0.f);
  }
  __syncthreads();
  if (tid < Cc) {
    float s = se_b2[tid];
#pragma unroll
    for (int h = 0; h < 3; ++h) s += se_w2[tid * 3 + h] * SE2[h];
    SEB[tid] = wf[0] / (1.f + __expf(-s));
  }
  // restage proj weights into (dead) qkv-weight region
  for (int i = tid; i < 48 * 64; i += 128) {
    const int o = i >> 6, k = i & 63;
    WB[i] = (_Float16)(k < Cc ? proj_w[o * Cc + k] : 0.f);
  }
  __syncthreads();

  // ---- out = attn + wf * xw * se  (f16 A-tile for proj) ----
  for (int i = tid; i < 64 * 64; i += 128) {
    const int m = i >> 6, k = i & 63;
    float v = 0.f;
    if (k < Cc) v = AO[m * Cc + k] + SEB[k] * (float)XW[m * 64 + k];
    A2[i] = (_Float16)v;
  }
  __syncthreads();

  // ---- proj GEMM + window-reverse + reverse shift + residual to xmid ----
  for (int t = wave; t < 12; t += 4) {
    const int mt = t & 3, nt = t >> 2;
    v8f acc = {};
#pragma unroll
    for (int ks = 0; ks < 2; ++ks) {
      const v16h a = ldA(A2 + (mt * 16) * 64 + ks * 32, 64);
      const v16h b = ldB(WB + (nt * 16) * 64 + ks * 32, 64);
      acc = WMMA_F16(a, b, acc);
    }
    const int n = lane & 15, mb = (lane >> 4) * 8;
    const int cg = nt * 16 + n;
    const float bias = proj_b[cg];
#pragma unroll
    for (int r = 0; r < 8; ++r) {
      const int m = mt * 16 + mb + r;
      const int rs = ((wh << 3) + (m >> 3) + SS) & (Hh - 1);
      const int cs = ((ww << 3) + (m & 7) + SS) & (Ww - 1);
      const size_t pos = ((size_t)bat * (Hh * Ww) + (size_t)rs * Ww + cs) * Cc + cg;
      xmid[pos] = x[pos] + acc[r] + bias;
    }
  }
}

// =====================================================================
// Kernel 2: LN2 + fc1 + GELU(erf) + fc2 + residual, 64 tokens / block.
// out and xmid alias (each element read+written by exactly one thread).
// =====================================================================
__global__ void __launch_bounds__(128)
swin_mlp_kernel(const float* __restrict__ xmid,
                const float* __restrict__ g2, const float* __restrict__ b2,
                const float* __restrict__ fc1_w, const float* __restrict__ fc1_b,
                const float* __restrict__ fc2_w, const float* __restrict__ fc2_b,
                float* __restrict__ out) {
  extern __shared__ char smem[];
  _Float16* A  = (_Float16*)(smem + OFF_MA);
  _Float16* W1 = (_Float16*)(smem + OFF_W1);
  _Float16* HB = (_Float16*)(smem + OFF_HB);
  _Float16* W2 = (_Float16*)(smem + OFF_W2);

  const int tid = threadIdx.x, wave = tid >> 5, lane = tid & 31;
  const size_t t0 = (size_t)blockIdx.x * 64;

  if (tid == 0) { __builtin_prefetch(fc1_w, 0, 1); __builtin_prefetch(fc2_w, 0, 1); }

  for (int i = tid; i < HID * 64; i += 128) {
    const int o = i >> 6, k = i & 63;
    W1[i] = (_Float16)(k < Cc ? fc1_w[o * Cc + k] : 0.f);
  }
  for (int i = tid; i < Cc * HID; i += 128) W2[i] = (_Float16)fc2_w[i];

  if (tid < 64) {
    const float* row = xmid + (t0 + tid) * Cc;
    float v[Cc];
    float mu = 0.f;
#pragma unroll
    for (int c = 0; c < Cc; ++c) { v[c] = row[c]; mu += v[c]; }
    mu *= (1.f / Cc);
    float s2 = 0.f;
#pragma unroll
    for (int c = 0; c < Cc; ++c) { const float d = v[c] - mu; s2 += d * d; }
    const float inv = rsqrtf(s2 * (1.f / Cc) + 1e-5f);
#pragma unroll
    for (int c = 0; c < Cc; ++c) A[tid * 64 + c] = (_Float16)((v[c] - mu) * inv * g2[c] + b2[c]);
#pragma unroll
    for (int c = Cc; c < 64; ++c) A[tid * 64 + c] = (_Float16)0.f;
  }
  __syncthreads();

  // fc1: [64x64] x [64x192], bias + exact GELU -> HB f16 [64][192]
  for (int t = wave; t < 48; t += 4) {
    const int mt = t & 3, nt = t >> 2;
    v8f acc = {};
#pragma unroll
    for (int ks = 0; ks < 2; ++ks) {
      const v16h a = ldA(A + (mt * 16) * 64 + ks * 32, 64);
      const v16h b = ldB(W1 + (nt * 16) * 64 + ks * 32, 64);
      acc = WMMA_F16(a, b, acc);
    }
    const int n = lane & 15, mb = (lane >> 4) * 8;
    const int ng = nt * 16 + n;
    const float bias = fc1_b[ng];
#pragma unroll
    for (int r = 0; r < 8; ++r) {
      float v = acc[r] + bias;
      v = 0.5f * v * (1.f + erff(v * 0.70710678f));
      HB[(mt * 16 + mb + r) * HID + ng] = (_Float16)v;
    }
  }
  __syncthreads();

  // fc2: [64x192] x [192x48] + bias + residual -> out
  for (int t = wave; t < 12; t += 4) {
    const int mt = t & 3, nt = t >> 2;
    v8f acc = {};
#pragma unroll
    for (int ks = 0; ks < 6; ++ks) {
      const v16h a = ldA(HB + (mt * 16) * HID + ks * 32, HID);
      const v16h b = ldB(W2 + (nt * 16) * HID + ks * 32, HID);
      acc = WMMA_F16(a, b, acc);
    }
    const int n = lane & 15, mb = (lane >> 4) * 8;
    const int cg = nt * 16 + n;
    const float bias = fc2_b[cg];
#pragma unroll
    for (int r = 0; r < 8; ++r) {
      const size_t idx = (t0 + mt * 16 + mb + r) * Cc + cg;
      out[idx] = xmid[idx] + acc[r] + bias;
    }
  }
}

// =====================================================================
extern "C" void kernel_launch(void* const* d_in, const int* in_sizes, int n_in,
                              void* d_out, int out_size, void* d_ws, size_t ws_size,
                              hipStream_t stream) {
  (void)in_sizes; (void)n_in; (void)out_size; (void)d_ws; (void)ws_size;

  const float* x        = (const float*)d_in[0];
  const float* gamma1   = (const float*)d_in[1];
  const float* beta1    = (const float*)d_in[2];
  const float* qkv_w    = (const float*)d_in[3];
  const float* qkv_b    = (const float*)d_in[4];
  const float* rpb      = (const float*)d_in[5];
  const float* wf       = (const float*)d_in[6];
  const float* se_w1    = (const float*)d_in[7];
  const float* se_b1    = (const float*)d_in[8];
  const float* se_w2    = (const float*)d_in[9];
  const float* se_b2    = (const float*)d_in[10];
  const float* proj_w   = (const float*)d_in[11];
  const float* proj_b   = (const float*)d_in[12];
  const float* gamma2   = (const float*)d_in[13];
  const float* beta2    = (const float*)d_in[14];
  const float* fc1_w    = (const float*)d_in[15];
  const float* fc1_b    = (const float*)d_in[16];
  const float* fc2_w    = (const float*)d_in[17];
  const float* fc2_b    = (const float*)d_in[18];
  float* xmid = (float*)d_out;   // mid residual lives in d_out, rewritten by mlp kernel

  (void)hipFuncSetAttribute((const void*)swin_attn_kernel,
                            hipFuncAttributeMaxDynamicSharedMemorySize, SMEM_ATTN);
  (void)hipFuncSetAttribute((const void*)swin_mlp_kernel,
                            hipFuncAttributeMaxDynamicSharedMemorySize, SMEM_MLP);

  const int nwin_blocks = Bb * NWIN;            // 8192 windows
  swin_attn_kernel<<<nwin_blocks, 128, SMEM_ATTN, stream>>>(
      x, gamma1, beta1, qkv_w, qkv_b, rpb, wf,
      se_w1, se_b1, se_w2, se_b2, proj_w, proj_b, xmid);

  const int nmlp_blocks = NTOK / 64;            // 8192 chunks of 64 tokens
  swin_mlp_kernel<<<nmlp_blocks, 128, SMEM_MLP, stream>>>(
      xmid, gamma2, beta2, fc1_w, fc1_b, fc2_w, fc2_b, (float*)d_out);
}
